// MixedSparseGatedMLP_75153337745538
// MI455X (gfx1250) — compile-verified, matrix operand order checked by reading, not run on hardware
//
#include <hip/hip_runtime.h>

typedef __attribute__((ext_vector_type(16))) __bf16 v16bf;
typedef __attribute__((ext_vector_type(8)))  float  v8f;
typedef __attribute__((ext_vector_type(4)))  int    v4i;

#define MTOT 4096   // B*S
#define HD   4096   // hidden
#define ID   11008  // intermediate

// CDNA5 async global->LDS copies (ASYNCcnt) when the toolchain exposes them.
#if defined(__AMDGCN__)
#if __has_builtin(__builtin_amdgcn_global_load_async_to_lds_b128) && \
    __has_builtin(__builtin_amdgcn_s_wait_asynccnt)
#define USE_ASYNC_COPY 1
#endif
#endif
#ifndef USE_ASYNC_COPY
#define USE_ASYNC_COPY 0
#endif

union FragU { v16bf v; uint4 q[2]; };

__device__ __forceinline__ v8f wmma_bf16(v16bf a, v16bf b, v8f c) {
  return __builtin_amdgcn_wmma_f32_16x16x32_bf16(false, a, false, b, (short)0, c, false, false);
}

// A/B 16-bit WMMA fragment from an LDS tile stored row-major, 32 bf16 per row.
// lane l: kk0=(l>>4)*8 -> K runs [kk0..kk0+7] and [16+kk0..16+kk0+7], each one b128.
__device__ __forceinline__ v16bf ldsFrag(const __bf16* base, int row, int lane) {
  const int kk0 = (lane >> 4) * 8;
  const __bf16* p = base + row * 32;
  FragU f;
  f.q[0] = *(const uint4*)(p + kk0);
  f.q[1] = *(const uint4*)(p + 16 + kk0);
  return f.v;
}

// One 16-byte global->LDS transfer. Async path: GLOBAL_LOAD_ASYNC_TO_LDS_B128,
// builtin signature (const v4i AS1*, v4i AS3*, imm offset, imm cpol).
// Pointer laundering: global generic == addrspace(1) value; LDS generic low 32
// bits == LDS byte offset (ISA 10.2 aperture mapping).
__device__ __forceinline__ void copy16(const __bf16* g, __bf16* l) {
#if USE_ASYNC_COPY
  __builtin_amdgcn_global_load_async_to_lds_b128(
      (__attribute__((address_space(1))) v4i*)(__UINTPTR_TYPE__)g,
      (__attribute__((address_space(3))) v4i*)(unsigned)(__UINTPTR_TYPE__)l,
      0, 0);
#else
  *(uint4*)l = *(const uint4*)g;
#endif
}

// Stage a 128x32 bf16 tile (rows row0.., cols k0..k0+31 of a [*, ld] matrix).
// 256 threads x 2 transfers of 8 bf16 -> fully coalesced b128s.
__device__ __forceinline__ void stageTile(__bf16* dst, const __bf16* __restrict__ src,
                                          int row0, int ld, int k0, int t) {
#pragma unroll
  for (int i = 0; i < 2; ++i) {
    int u = t + i * 256;
    int row = u >> 2;
    int chunk = (u & 3) * 8;
    copy16(src + (size_t)(row0 + row) * ld + k0 + chunk, dst + row * 32 + chunk);
  }
}

// Wait for this wave's outstanding async copies, then workgroup barrier.
__device__ __forceinline__ void stageWait() {
#if USE_ASYNC_COPY
  __builtin_amdgcn_s_wait_asynccnt(0);
#endif
  __syncthreads();
}

// ---------------------------------------------------------------------------
// fp32 -> bf16 elementwise (n divisible by 4)
__global__ void __launch_bounds__(256) cvt_bf16_kernel(const float* __restrict__ src,
                                                       __bf16* __restrict__ dst, int n4) {
  int i = blockIdx.x * 256 + threadIdx.x;
  if (i >= n4) return;
  float4 v = *(const float4*)(src + (size_t)i * 4);
  union { uint2 u; __bf16 h[4]; } w;
  w.h[0] = (__bf16)v.x; w.h[1] = (__bf16)v.y; w.h[2] = (__bf16)v.z; w.h[3] = (__bf16)v.w;
  *(uint2*)(dst + (size_t)i * 4) = w.u;
}

// fp32 [K][N] -> bf16 [N][K]  (tiled 32x32 transpose through LDS)
__global__ void __launch_bounds__(256) transpose_cvt_kernel(const float* __restrict__ src,
                                                            __bf16* __restrict__ dst,
                                                            int K, int N) {
  __shared__ float tile[32][33];
  const int t = threadIdx.x;
  const int tx = t & 31, ty = t >> 5;
  const int n0 = blockIdx.x * 32;
  const int k0 = blockIdx.y * 32;
#pragma unroll
  for (int i = 0; i < 4; ++i) {
    int r = ty + i * 8;
    tile[r][tx] = src[(size_t)(k0 + r) * N + n0 + tx];
  }
  __syncthreads();
#pragma unroll
  for (int i = 0; i < 4; ++i) {
    int r = ty + i * 8;
    dst[(size_t)(n0 + r) * K + k0 + tx] = (__bf16)tile[tx][r];
  }
}

// lora_b fp32 [16][N] -> bf16 [N][32] (zero-padded K 16->32), N-major
__global__ void __launch_bounds__(256) lorab_cvt_kernel(const float* __restrict__ src,
                                                        __bf16* __restrict__ dst, int N) {
  int n = blockIdx.x * 256 + threadIdx.x;
  if (n >= N) return;
#pragma unroll
  for (int r = 0; r < 16; ++r) dst[(size_t)n * 32 + r] = (__bf16)src[(size_t)r * N + n];
#pragma unroll
  for (int r = 16; r < 32; ++r) dst[(size_t)n * 32 + r] = (__bf16)0.0f;
}

// T = X(bf16 [M][K]) @ A(fp32 [K][16]) -> bf16 [M][32] zero-padded. One block per row.
__global__ void __launch_bounds__(256) lora_t_kernel(const __bf16* __restrict__ X,
                                                     const float* __restrict__ A,
                                                     __bf16* __restrict__ T, int K) {
  __shared__ float sacc[256 * 16];
  const int t = threadIdx.x;
  const int m = blockIdx.x;
  float acc[16];
#pragma unroll
  for (int r = 0; r < 16; ++r) acc[r] = 0.0f;
  const __bf16* xrow = X + (size_t)m * K;
  for (int k = t; k < K; k += 256) {
    float x = (float)xrow[k];
    const float4* a4 = (const float4*)(A + (size_t)k * 16);
#pragma unroll
    for (int q = 0; q < 4; ++q) {
      float4 a = a4[q];
      acc[q * 4 + 0] += x * a.x;
      acc[q * 4 + 1] += x * a.y;
      acc[q * 4 + 2] += x * a.z;
      acc[q * 4 + 3] += x * a.w;
    }
  }
#pragma unroll
  for (int r = 0; r < 16; ++r) sacc[t * 16 + r] = acc[r];
  __syncthreads();
  if (t < 16) {
    float s = 0.0f;
    for (int i = 0; i < 256; ++i) s += sacc[i * 16 + t];
    T[(size_t)m * 32 + t] = (__bf16)s;
  } else if (t < 32) {
    T[(size_t)m * 32 + t] = (__bf16)0.0f;
  }
}

// ---------------------------------------------------------------------------
// Fused gate/up GEMM + LoRA K-step + relu(g)*u, writes x3 bf16 [M][I].
// 128x128 block tile, 8 waves (4m x 2n), wave tile 32x64. Double-buffered LDS:
// async-copy panel kk+1 while WMMA consumes panel kk; one barrier per K-step.
__global__ void __launch_bounds__(256)
gateup_gemm_kernel(const __bf16* __restrict__ xb, const __bf16* __restrict__ wgT,
                   const __bf16* __restrict__ wuT, const __bf16* __restrict__ Tg,
                   const __bf16* __restrict__ Tu, const __bf16* __restrict__ BgT,
                   const __bf16* __restrict__ BuT, __bf16* __restrict__ x3b) {
  __shared__ __align__(16) __bf16 sA[2][128 * 32];
  __shared__ __align__(16) __bf16 sB0[2][128 * 32];
  __shared__ __align__(16) __bf16 sB1[2][128 * 32];

  const int t = threadIdx.x;
  const int lane = t & 31;
  const int wave = t >> 5;
  const int wm = wave >> 1;  // 0..3 -> 32 rows each
  const int wn = wave & 1;   // 0..1 -> 64 cols each
  const int m0 = blockIdx.x * 128;
  const int n0 = blockIdx.y * 128;
  const int NK = HD / 32;

  v8f accG[2][4] = {};
  v8f accU[2][4] = {};

  // prologue: panel 0
  stageTile(sA[0], xb, m0, HD, 0, t);
  stageTile(sB0[0], wgT, n0, HD, 0, t);
  stageTile(sB1[0], wuT, n0, HD, 0, t);
  stageWait();

  for (int kk = 0; kk < NK; ++kk) {
    const int cur = kk & 1, nxt = cur ^ 1;
    if (kk + 1 < NK) {
      const int k1 = (kk + 1) * 32;
      stageTile(sA[nxt], xb, m0, HD, k1, t);
      stageTile(sB0[nxt], wgT, n0, HD, k1, t);
      stageTile(sB1[nxt], wuT, n0, HD, k1, t);
    }
    if (kk + 2 < NK) {  // pull the panel after next toward L2
      const int k2 = (kk + 2) * 32;
      __builtin_prefetch(xb + (size_t)(m0 + (t >> 1)) * HD + k2, 0, 1);
      __builtin_prefetch(wgT + (size_t)(n0 + (t >> 1)) * HD + k2, 0, 1);
      __builtin_prefetch(wuT + (size_t)(n0 + (t >> 1)) * HD + k2, 0, 1);
    }
    v16bf a0 = ldsFrag(sA[cur], wm * 32 + (lane & 15), lane);
    v16bf a1 = ldsFrag(sA[cur], wm * 32 + 16 + (lane & 15), lane);
#pragma unroll
    for (int j = 0; j < 4; ++j) {
      const int nrow = wn * 64 + j * 16 + (lane & 15);
      v16bf bg = ldsFrag(sB0[cur], nrow, lane);
      v16bf bu = ldsFrag(sB1[cur], nrow, lane);
      accG[0][j] = wmma_bf16(a0, bg, accG[0][j]);
      accG[1][j] = wmma_bf16(a1, bg, accG[1][j]);
      accU[0][j] = wmma_bf16(a0, bu, accU[0][j]);
      accU[1][j] = wmma_bf16(a1, bu, accU[1][j]);
    }
    stageWait();
  }

  // LoRA: one extra zero-padded K=32 step with A = T, B = lora_b^T
  stageTile(sA[0], Tg, m0, 32, 0, t);
  stageTile(sA[1], Tu, m0, 32, 0, t);
  stageTile(sB0[0], BgT, n0, 32, 0, t);
  stageTile(sB1[0], BuT, n0, 32, 0, t);
  stageWait();
  {
    v16bf ag0 = ldsFrag(sA[0], wm * 32 + (lane & 15), lane);
    v16bf ag1 = ldsFrag(sA[0], wm * 32 + 16 + (lane & 15), lane);
    v16bf au0 = ldsFrag(sA[1], wm * 32 + (lane & 15), lane);
    v16bf au1 = ldsFrag(sA[1], wm * 32 + 16 + (lane & 15), lane);
#pragma unroll
    for (int j = 0; j < 4; ++j) {
      const int nrow = wn * 64 + j * 16 + (lane & 15);
      v16bf bg = ldsFrag(sB0[0], nrow, lane);
      v16bf bu = ldsFrag(sB1[0], nrow, lane);
      accG[0][j] = wmma_bf16(ag0, bg, accG[0][j]);
      accG[1][j] = wmma_bf16(ag1, bg, accG[1][j]);
      accU[0][j] = wmma_bf16(au0, bu, accU[0][j]);
      accU[1][j] = wmma_bf16(au1, bu, accU[1][j]);
    }
  }

  // x3 = relu(g) * u -> bf16. C/D layout: m = v + 8*(lane>>4), n = lane&15.
#pragma unroll
  for (int i = 0; i < 2; ++i)
#pragma unroll
    for (int j = 0; j < 4; ++j) {
      const int n = n0 + wn * 64 + j * 16 + (lane & 15);
#pragma unroll
      for (int v = 0; v < 8; ++v) {
        const int m = m0 + wm * 32 + i * 16 + v + 8 * (lane >> 4);
        float g = accG[i][j][v];
        float u = accU[i][j][v];
        float r = (g > 0.0f ? g : 0.0f) * u;
        x3b[(size_t)m * ID + n] = (__bf16)r;
      }
    }
}

// Down GEMM + LoRA K-step; fp32 output [M][H]. Same double-buffered skeleton.
__global__ void __launch_bounds__(256)
down_gemm_kernel(const __bf16* __restrict__ x3b, const __bf16* __restrict__ wdT,
                 const __bf16* __restrict__ Td, const __bf16* __restrict__ BdT,
                 float* __restrict__ out) {
  __shared__ __align__(16) __bf16 sA[2][128 * 32];
  __shared__ __align__(16) __bf16 sB[2][128 * 32];

  const int t = threadIdx.x;
  const int lane = t & 31;
  const int wave = t >> 5;
  const int wm = wave >> 1;
  const int wn = wave & 1;
  const int m0 = blockIdx.x * 128;
  const int n0 = blockIdx.y * 128;
  const int NK = ID / 32;

  v8f acc[2][4] = {};

  stageTile(sA[0], x3b, m0, ID, 0, t);
  stageTile(sB[0], wdT, n0, ID, 0, t);
  stageWait();

  for (int kk = 0; kk < NK; ++kk) {
    const int cur = kk & 1, nxt = cur ^ 1;
    if (kk + 1 < NK) {
      const int k1 = (kk + 1) * 32;
      stageTile(sA[nxt], x3b, m0, ID, k1, t);
      stageTile(sB[nxt], wdT, n0, ID, k1, t);
    }
    if (kk + 2 < NK) {
      const int k2 = (kk + 2) * 32;
      __builtin_prefetch(x3b + (size_t)(m0 + (t >> 1)) * ID + k2, 0, 1);
      __builtin_prefetch(wdT + (size_t)(n0 + (t >> 1)) * ID + k2, 0, 1);
    }
    v16bf a0 = ldsFrag(sA[cur], wm * 32 + (lane & 15), lane);
    v16bf a1 = ldsFrag(sA[cur], wm * 32 + 16 + (lane & 15), lane);
#pragma unroll
    for (int j = 0; j < 4; ++j) {
      const int nrow = wn * 64 + j * 16 + (lane & 15);
      v16bf b = ldsFrag(sB[cur], nrow, lane);
      acc[0][j] = wmma_bf16(a0, b, acc[0][j]);
      acc[1][j] = wmma_bf16(a1, b, acc[1][j]);
    }
    stageWait();
  }

  // LoRA down K-step
  stageTile(sA[0], Td, m0, 32, 0, t);
  stageTile(sB[0], BdT, n0, 32, 0, t);
  stageWait();
  {
    v16bf a0 = ldsFrag(sA[0], wm * 32 + (lane & 15), lane);
    v16bf a1 = ldsFrag(sA[0], wm * 32 + 16 + (lane & 15), lane);
#pragma unroll
    for (int j = 0; j < 4; ++j) {
      const int nrow = wn * 64 + j * 16 + (lane & 15);
      v16bf b = ldsFrag(sB[0], nrow, lane);
      acc[0][j] = wmma_bf16(a0, b, acc[0][j]);
      acc[1][j] = wmma_bf16(a1, b, acc[1][j]);
    }
  }

#pragma unroll
  for (int i = 0; i < 2; ++i)
#pragma unroll
    for (int j = 0; j < 4; ++j) {
      const int n = n0 + wn * 64 + j * 16 + (lane & 15);
#pragma unroll
      for (int v = 0; v < 8; ++v) {
        const int m = m0 + wm * 32 + i * 16 + v + 8 * (lane >> 4);
        out[(size_t)m * HD + n] = acc[i][j][v];
      }
    }
}

// ---------------------------------------------------------------------------
extern "C" void kernel_launch(void* const* d_in, const int* in_sizes, int n_in,
                              void* d_out, int out_size, void* d_ws, size_t ws_size,
                              hipStream_t stream) {
  (void)in_sizes; (void)n_in; (void)out_size; (void)ws_size;
  const float* x1     = (const float*)d_in[0];
  const float* w_gate = (const float*)d_in[1];
  const float* a_gate = (const float*)d_in[2];
  const float* b_gate = (const float*)d_in[3];
  const float* w_up   = (const float*)d_in[4];
  const float* a_up   = (const float*)d_in[5];
  const float* b_up   = (const float*)d_in[6];
  const float* w_down = (const float*)d_in[7];
  const float* a_down = (const float*)d_in[8];
  const float* b_down = (const float*)d_in[9];
  float* out = (float*)d_out;

  char* ws = (char*)d_ws;
  size_t off = 0;
  auto alloc = [&](size_t bytes) -> void* {
    void* p = ws + off;
    off += (bytes + 255) & ~(size_t)255;
    return p;
  };
  __bf16* xb  = (__bf16*)alloc((size_t)MTOT * HD * 2);
  __bf16* wgT = (__bf16*)alloc((size_t)ID * HD * 2);
  __bf16* wuT = (__bf16*)alloc((size_t)ID * HD * 2);
  __bf16* wdT = (__bf16*)alloc((size_t)HD * ID * 2);
  __bf16* x3b = (__bf16*)alloc((size_t)MTOT * ID * 2);
  __bf16* Tg  = (__bf16*)alloc((size_t)MTOT * 32 * 2);
  __bf16* Tu  = (__bf16*)alloc((size_t)MTOT * 32 * 2);
  __bf16* Td  = (__bf16*)alloc((size_t)MTOT * 32 * 2);
  __bf16* BgT = (__bf16*)alloc((size_t)ID * 32 * 2);
  __bf16* BuT = (__bf16*)alloc((size_t)ID * 32 * 2);
  __bf16* BdT = (__bf16*)alloc((size_t)HD * 32 * 2);

  // 1) x1 -> bf16
  cvt_bf16_kernel<<<(MTOT * HD / 4 + 255) / 256, 256, 0, stream>>>(x1, xb, MTOT * HD / 4);
  // 2) weights -> bf16, transposed to N-major
  transpose_cvt_kernel<<<dim3(ID / 32, HD / 32), 256, 0, stream>>>(w_gate, wgT, HD, ID);
  transpose_cvt_kernel<<<dim3(ID / 32, HD / 32), 256, 0, stream>>>(w_up,   wuT, HD, ID);
  transpose_cvt_kernel<<<dim3(HD / 32, ID / 32), 256, 0, stream>>>(w_down, wdT, ID, HD);
  // 3) lora_b -> bf16 [N][32] padded
  lorab_cvt_kernel<<<(ID + 255) / 256, 256, 0, stream>>>(b_gate, BgT, ID);
  lorab_cvt_kernel<<<(ID + 255) / 256, 256, 0, stream>>>(b_up,   BuT, ID);
  lorab_cvt_kernel<<<(HD + 255) / 256, 256, 0, stream>>>(b_down, BdT, HD);
  // 4) T = x @ lora_a (gate, up)
  lora_t_kernel<<<MTOT, 256, 0, stream>>>(xb, a_gate, Tg, HD);
  lora_t_kernel<<<MTOT, 256, 0, stream>>>(xb, a_up,   Tu, HD);
  // 5) fused gate/up + relu*mul -> x3 (bf16)
  gateup_gemm_kernel<<<dim3(MTOT / 128, ID / 128), 256, 0, stream>>>(
      xb, wgT, wuT, Tg, Tu, BgT, BuT, x3b);
  // 6) T_down = x3 @ lora_a_down
  lora_t_kernel<<<MTOT, 256, 0, stream>>>(x3b, a_down, Td, ID);
  // 7) down projection -> out (fp32)
  down_gemm_kernel<<<dim3(MTOT / 128, HD / 128), 256, 0, stream>>>(x3b, wdT, Td, BdT, out);
}
